// TransformerBlock_27144193311234
// MI455X (gfx1250) — compile-verified, matrix operand order checked by reading, not run on hardware
//
#include <hip/hip_runtime.h>
#include <hip/hip_bf16.h>
#include <stdint.h>

#define D_MODEL 1024
#define N_HEADS 16
#define DEPTH   64
#define DFF     4096
#define BATCH   2
#define SEQ     2048
#define ROWS    (BATCH * SEQ)   // 4096

typedef __bf16 bf16_t;
typedef __attribute__((ext_vector_type(16))) __bf16 v16bf;
typedef __attribute__((ext_vector_type(8)))  float  v8f;
typedef int v4i_vs __attribute__((vector_size(4 * sizeof(int))));   // builtin's pointee type

#if defined(__has_builtin)
#if __has_builtin(__builtin_amdgcn_global_load_async_to_lds_b128) && \
    __has_builtin(__builtin_amdgcn_s_wait_asynccnt)
#define HAVE_ASYNC_LDS 1
#endif
#endif

#ifdef HAVE_ASYNC_LDS
// 16B global->LDS async copy (GLOBAL_LOAD_ASYNC_TO_LDS_B128, ASYNCcnt).
// Builtin takes int4-vector pointers: global operand in AS1, LDS operand in AS3.
// AS1 value == generic global address; AS3 value == generic LDS addr[31:0].
__device__ __forceinline__ void async_copy_b128(const void* g, void* l) {
  __builtin_amdgcn_global_load_async_to_lds_b128(
      (__attribute__((address_space(1))) v4i_vs*)(uintptr_t)g,
      (__attribute__((address_space(3))) v4i_vs*)(uint32_t)(uintptr_t)l,
      0, 0);
}
#endif

// 16-bit A-matrix 16x32 K index pattern (ISA 7.12.2):
// lanes 0-15 hold K=0..7 / 16..23; lanes 16-31 hold K=8..15 / 24..31
__device__ __forceinline__ int kpat(int half, int i) {
  return (i < 8) ? (half * 8 + i) : (16 + half * 8 + (i - 8));
}

// ---------------------------------------------------------------------------
// fp32 -> bf16 bulk conversion (weights / once per launch)
// ---------------------------------------------------------------------------
__global__ __launch_bounds__(256) void cvt_f32_bf16(const float* __restrict__ in,
                                                    bf16_t* __restrict__ out,
                                                    int n4) {
  const int i = blockIdx.x * 256 + threadIdx.x;
  if (i < n4) {
    const float4 v = ((const float4*)in)[i];
    union { bf16_t h[4]; uint2 u; } o;
    o.h[0] = (bf16_t)v.x; o.h[1] = (bf16_t)v.y;
    o.h[2] = (bf16_t)v.z; o.h[3] = (bf16_t)v.w;
    ((uint2*)out)[i] = o.u;
  }
}

// ---------------------------------------------------------------------------
// GEMM: C[M,N] = act(A[M,K] @ B[K,N] + bias[N]); A fp32, B bf16, C fp32/bf16.
// Block: 256 threads = 8 waves; block tile 64(M) x 128(N); wave tile 32x32
// (2x2 WMMA register blocking); K stepped by 32. B staged via async LDS copy.
// ---------------------------------------------------------------------------
template <bool RELU, typename CT>
__global__ __launch_bounds__(256) void gemm_bf16(const float*  __restrict__ A,
                                                 const bf16_t* __restrict__ B,
                                                 const float*  __restrict__ bias,
                                                 CT* __restrict__ C,
                                                 int M, int N, int K) {
  __shared__ __align__(16) bf16_t sA[64][32];    // 4 KB
  __shared__ __align__(16) bf16_t sB[32][128];   // 8 KB

  const int tid    = threadIdx.x;
  const int lane   = tid & 31;
  const int wid    = tid >> 5;
  const int wm     = wid >> 2;   // 0..1 : wave row tile (32 rows)
  const int wn     = wid & 3;    // 0..3 : wave col tile (32 cols)
  const int lane15 = lane & 15;
  const int half   = lane >> 4;

  const int m0 = blockIdx.y * 64;
  const int n0 = blockIdx.x * 128;

  // A staging coords: 64 rows x 32 k, 8 floats/thread
  const int arow = tid >> 2;          // 0..63
  const int acol = (tid & 3) * 8;     // 0,8,16,24

  v8f acc[2][2] = {};

  const int ksteps = K >> 5;
  for (int kt = 0; kt < ksteps; ++kt) {
    const int k0 = kt << 5;

    // speculative prefetch two K-steps ahead (global_prefetch_b8)
    __builtin_prefetch(&A[(size_t)(m0 + arow) * K + (k0 + 64) + acol], 0, 1);
    __builtin_prefetch(&B[(size_t)(k0 + 64 + (tid >> 4)) * N + n0], 0, 1);

    __syncthreads();

    // ---- B tile (bf16 bytes): 32x128 = 512 x 16B chunks, 2 per thread
#pragma unroll
    for (int c = 0; c < 2; ++c) {
      const int ch   = tid + c * 256;   // 0..511
      const int brow = ch >> 4;         // 0..31
      const int bc8  = (ch & 15) * 8;   // bf16 column
      const void* src = &B[(size_t)(k0 + brow) * N + n0 + bc8];
#ifdef HAVE_ASYNC_LDS
      async_copy_b128(src, &sB[brow][bc8]);
#else
      *(uint4*)&sB[brow][bc8] = *(const uint4*)src;
#endif
    }

    // ---- A tile (fp32 -> bf16 convert while staging)
    {
      const float4 a0 = *(const float4*)&A[(size_t)(m0 + arow) * K + k0 + acol];
      const float4 a1 = *(const float4*)&A[(size_t)(m0 + arow) * K + k0 + acol + 4];
      sA[arow][acol + 0] = (bf16_t)a0.x;  sA[arow][acol + 1] = (bf16_t)a0.y;
      sA[arow][acol + 2] = (bf16_t)a0.z;  sA[arow][acol + 3] = (bf16_t)a0.w;
      sA[arow][acol + 4] = (bf16_t)a1.x;  sA[arow][acol + 5] = (bf16_t)a1.y;
      sA[arow][acol + 6] = (bf16_t)a1.z;  sA[arow][acol + 7] = (bf16_t)a1.w;
    }

#ifdef HAVE_ASYNC_LDS
    __builtin_amdgcn_s_wait_asynccnt(0);
#endif
    __syncthreads();

    v16bf af[2], bfv[2];
#pragma unroll
    for (int s = 0; s < 2; ++s)
#pragma unroll
      for (int i = 0; i < 16; ++i) {
        const int kk = kpat(half, i);
        af[s][i]  = sA[wm * 32 + s * 16 + lane15][kk];
        bfv[s][i] = sB[kk][wn * 32 + s * 16 + lane15];
      }
#pragma unroll
    for (int si = 0; si < 2; ++si)
#pragma unroll
      for (int sj = 0; sj < 2; ++sj)
        acc[si][sj] = __builtin_amdgcn_wmma_f32_16x16x32_bf16(
            false, af[si], false, bfv[sj], (short)0, acc[si][sj], false, false);
  }

  // epilogue: bias (+ReLU); C/D layout: slot r -> row r+8*half, col = lane15
#pragma unroll
  for (int si = 0; si < 2; ++si)
#pragma unroll
    for (int sj = 0; sj < 2; ++sj)
#pragma unroll
      for (int r = 0; r < 8; ++r) {
        const int row = m0 + wm * 32 + si * 16 + r + 8 * half;
        const int col = n0 + wn * 32 + sj * 16 + lane15;
        float v = acc[si][sj][r] + bias[col];
        if (RELU) v = fmaxf(v, 0.0f);
        C[(size_t)row * N + col] = (CT)v;
      }
}

// ---------------------------------------------------------------------------
// Causal attention, flash style. Block = 8 waves covering 8 consecutive
// 16-query tiles of one (b,h). Q/K/V are bf16 [B*S, D_MODEL]; K/V chunks
// async-staged into LDS; CTX written fp32.
// ---------------------------------------------------------------------------
__global__ __launch_bounds__(256) void attention_causal(const bf16_t* __restrict__ Q,
                                                        const bf16_t* __restrict__ Kb,
                                                        const bf16_t* __restrict__ Vb,
                                                        float* __restrict__ CTX) {
  __shared__ __align__(16) bf16_t sK[32][DEPTH];      // 4 KB
  __shared__ __align__(16) bf16_t sV[32][DEPTH];      // 4 KB
  __shared__ __align__(16) bf16_t plds[8][16 * 32];   // 8 KB per-wave P staging

  const int tid    = threadIdx.x;
  const int lane   = tid & 31;
  const int wid    = tid >> 5;
  const int lane15 = lane & 15;
  const int half   = lane >> 4;

  // block -> (b, h, query-tile group of 8)
  const int group = blockIdx.x >> 4;        // 0..31 : (b,h)
  const int qg    = blockIdx.x & 15;        // 0..15
  const int h     = group & (N_HEADS - 1);
  const int b     = group >> 4;
  const int qt    = qg * 8 + wid;           // this wave's query tile (0..127)

  const int   base_row = b * SEQ;
  const int   hofs     = h * DEPTH;
  const float scale    = 0.125f;   // 1/sqrt(64)

  // staging coords: 32 rows x 64 bf16 cols = 256 x 16B chunks, 1 per thread
  const int srow = tid >> 3;        // 0..31
  const int sc8  = (tid & 7) * 8;   // bf16 column

  // Q fragments: 16x64 as two 16x32 A-fragments, straight bf16 loads
  v16bf qf[2];
  {
    const bf16_t* qptr = Q + (size_t)(base_row + qt * 16 + lane15) * D_MODEL + hofs;
#pragma unroll
    for (int dc = 0; dc < 2; ++dc)
#pragma unroll
      for (int i = 0; i < 16; ++i)
        qf[dc][i] = qptr[dc * 32 + kpat(half, i)];
  }

  v8f   acc[4] = {};
  float m_i[8], l_i[8];
#pragma unroll
  for (int r = 0; r < 8; ++r) { m_i[r] = -3.0e38f; l_i[r] = 0.0f; }

  const int lastk       = qt * 16 + 15;   // this wave's last valid key
  const int block_kcmax = qg * 4 + 3;     // block-uniform chunk bound

  bf16_t* pl = &plds[wid][0];

  for (int kc = 0; kc <= block_kcmax; ++kc) {
    const int kb = kc << 5;

    // ---- async staging of K/V chunk into LDS (coalesced b128)
    __syncthreads();
    {
      const void* ksrc = &Kb[(size_t)(base_row + kb + srow) * D_MODEL + hofs + sc8];
      const void* vsrc = &Vb[(size_t)(base_row + kb + srow) * D_MODEL + hofs + sc8];
#ifdef HAVE_ASYNC_LDS
      async_copy_b128(ksrc, &sK[srow][sc8]);
      async_copy_b128(vsrc, &sV[srow][sc8]);
      __builtin_amdgcn_s_wait_asynccnt(0);
#else
      *(uint4*)&sK[srow][sc8] = *(const uint4*)ksrc;
      *(uint4*)&sV[srow][sc8] = *(const uint4*)vsrc;
#endif
    }
    __syncthreads();

    // wave-uniform causal skip: this wave needs no keys from this chunk
    if (kb > lastk) continue;

    // ---- scores: two 16-key subtiles, depth=64 -> 2 WMMA each
    v8f s[2];
#pragma unroll
    for (int st = 0; st < 2; ++st) {
      v8f z = {};
#pragma unroll
      for (int dc = 0; dc < 2; ++dc) {
        v16bf kf;
#pragma unroll
        for (int i = 0; i < 16; ++i)
          kf[i] = sK[st * 16 + lane15][dc * 32 + kpat(half, i)];
        z = __builtin_amdgcn_wmma_f32_16x16x32_bf16(false, qf[dc], false, kf,
                                                    (short)0, z, false, false);
      }
      s[st] = z;
    }

    // ---- online softmax over this 32-key chunk (width-16 lane reductions)
    float alph[8];
#pragma unroll
    for (int r = 0; r < 8; ++r) {
      const int qrow = qt * 16 + r + 8 * half;
      const int key0 = kb + lane15;
      const int key1 = kb + 16 + lane15;
      float a0 = s[0][r] * scale + (key0 > qrow ? -1.0e9f : 0.0f);
      float a1 = s[1][r] * scale + (key1 > qrow ? -1.0e9f : 0.0f);
      float cm = fmaxf(a0, a1);
#pragma unroll
      for (int off = 8; off > 0; off >>= 1)
        cm = fmaxf(cm, __shfl_xor(cm, off, 16));
      const float mnew  = fmaxf(m_i[r], cm);
      const float alpha = __expf(m_i[r] - mnew);
      m_i[r] = mnew;
      const float e0 = __expf(a0 - mnew);
      const float e1 = __expf(a1 - mnew);
      float rs = e0 + e1;
#pragma unroll
      for (int off = 8; off > 0; off >>= 1)
        rs += __shfl_xor(rs, off, 16);
      l_i[r]  = l_i[r] * alpha + rs;
      alph[r] = alpha;
      // stash P (C layout) into per-wave LDS tile for A-layout reload
      const int prow = r + 8 * half;
      pl[prow * 32 + lane15]      = (bf16_t)e0;
      pl[prow * 32 + 16 + lane15] = (bf16_t)e1;
    }

    // rescale running context
#pragma unroll
    for (int nt = 0; nt < 4; ++nt)
#pragma unroll
      for (int r = 0; r < 8; ++r)
        acc[nt][r] *= alph[r];

    // ---- P fragment (16x32 A-layout); same-wave DS ops are in-order
    v16bf pf;
#pragma unroll
    for (int i = 0; i < 16; ++i)
      pf[i] = pl[lane15 * 32 + kpat(half, i)];

    // ---- PV: P(16x32) x V(32x64) -> four 16x16 WMMA
#pragma unroll
    for (int nt = 0; nt < 4; ++nt) {
      v16bf vf;
#pragma unroll
      for (int i = 0; i < 16; ++i)
        vf[i] = sV[kpat(half, i)][nt * 16 + lane15];
      acc[nt] = __builtin_amdgcn_wmma_f32_16x16x32_bf16(false, pf, false, vf,
                                                        (short)0, acc[nt],
                                                        false, false);
    }
  }

  // ---- normalize and write ctx (fp32) into [B*S, D_MODEL] layout
#pragma unroll
  for (int nt = 0; nt < 4; ++nt)
#pragma unroll
    for (int r = 0; r < 8; ++r) {
      const int row = base_row + qt * 16 + r + 8 * half;
      CTX[(size_t)row * D_MODEL + hofs + nt * 16 + lane15] = acc[nt][r] / l_i[r];
    }
}

// ---------------------------------------------------------------------------
// Fused residual add + LayerNorm over D_MODEL=1024. One block per row.
// ---------------------------------------------------------------------------
__global__ __launch_bounds__(256) void add_layernorm(const float* __restrict__ A,
                                                     const float* __restrict__ Bres,
                                                     const float* __restrict__ gamma,
                                                     const float* __restrict__ beta,
                                                     float* __restrict__ Out) {
  __shared__ float red[16];
  const int row  = blockIdx.x;
  const int tid  = threadIdx.x;
  const int lane = tid & 31;
  const int wid  = tid >> 5;
  const int e    = tid * 4;

  const float4 va = *(const float4*)&A[(size_t)row * D_MODEL + e];
  const float4 vb = *(const float4*)&Bres[(size_t)row * D_MODEL + e];
  const float x0 = va.x + vb.x, x1 = va.y + vb.y;
  const float x2 = va.z + vb.z, x3 = va.w + vb.w;

  float s = x0 + x1 + x2 + x3;
  float q = x0 * x0 + x1 * x1 + x2 * x2 + x3 * x3;
#pragma unroll
  for (int off = 16; off > 0; off >>= 1) {
    s += __shfl_xor(s, off, 32);
    q += __shfl_xor(q, off, 32);
  }
  if (lane == 0) { red[wid] = s; red[8 + wid] = q; }
  __syncthreads();
  if (tid == 0) {
    float ts = 0.0f, tq = 0.0f;
#pragma unroll
    for (int w = 0; w < 8; ++w) { ts += red[w]; tq += red[8 + w]; }
    const float mean = ts * (1.0f / D_MODEL);
    const float var  = tq * (1.0f / D_MODEL) - mean * mean;
    red[0] = mean;
    red[1] = rsqrtf(var + 1e-6f);
  }
  __syncthreads();
  const float mean = red[0];
  const float inv  = red[1];

  const float4 g  = *(const float4*)&gamma[e];
  const float4 be = *(const float4*)&beta[e];
  float4 y;
  y.x = (x0 - mean) * inv * g.x + be.x;
  y.y = (x1 - mean) * inv * g.y + be.y;
  y.z = (x2 - mean) * inv * g.z + be.z;
  y.w = (x3 - mean) * inv * g.w + be.w;
  *(float4*)&Out[(size_t)row * D_MODEL + e] = y;
}

// ---------------------------------------------------------------------------
// Launch: full transformer block pipeline.
// ---------------------------------------------------------------------------
extern "C" void kernel_launch(void* const* d_in, const int* in_sizes, int n_in,
                              void* d_out, int out_size, void* d_ws,
                              size_t ws_size, hipStream_t stream) {
  const float* x   = (const float*)d_in[0];
  // d_in[1] = dense causal mask: unused, causality computed inline
  const float* wq  = (const float*)d_in[2];
  const float* bq  = (const float*)d_in[3];
  const float* wk  = (const float*)d_in[4];
  const float* bk  = (const float*)d_in[5];
  const float* wv  = (const float*)d_in[6];
  const float* bv  = (const float*)d_in[7];
  const float* wo  = (const float*)d_in[8];
  const float* bo  = (const float*)d_in[9];
  const float* g1  = (const float*)d_in[10];
  const float* be1 = (const float*)d_in[11];
  const float* w1  = (const float*)d_in[12];
  const float* b1  = (const float*)d_in[13];
  const float* w2  = (const float*)d_in[14];
  const float* b2  = (const float*)d_in[15];
  const float* g2  = (const float*)d_in[16];
  const float* be2 = (const float*)d_in[17];
  float* out = (float*)d_out;

  const size_t NE  = (size_t)ROWS * D_MODEL;     // 4M
  const size_t WSQ = (size_t)D_MODEL * D_MODEL;  // 1M
  const size_t WFF = (size_t)D_MODEL * DFF;      // 4M

  bf16_t* WQb = (bf16_t*)d_ws;
  bf16_t* WKb = WQb + WSQ;
  bf16_t* WVb = WKb + WSQ;
  bf16_t* WOb = WVb + WSQ;
  bf16_t* W1b = WOb + WSQ;
  bf16_t* W2b = W1b + WFF;
  bf16_t* Qbf = W2b + WFF;
  bf16_t* Kbf = Qbf + NE;
  bf16_t* Vbf = Kbf + NE;
  float*  CTX = (float*)(Vbf + NE);   // 48 MB of bf16 precede (16B aligned)
  float*  ATT = CTX + NE;
  float*  O1  = ATT + NE;
  float*  H1  = O1 + NE;              // ROWS x DFF
  float*  FFN = CTX;                  // reuse: CTX dead after output proj

  const dim3 blk(256);

  // ---- one-time (per launch) fp32 -> bf16 weight conversion
  cvt_f32_bf16<<<dim3((int)(WSQ / 4 / 256)), blk, 0, stream>>>(wq, WQb, (int)(WSQ / 4));
  cvt_f32_bf16<<<dim3((int)(WSQ / 4 / 256)), blk, 0, stream>>>(wk, WKb, (int)(WSQ / 4));
  cvt_f32_bf16<<<dim3((int)(WSQ / 4 / 256)), blk, 0, stream>>>(wv, WVb, (int)(WSQ / 4));
  cvt_f32_bf16<<<dim3((int)(WSQ / 4 / 256)), blk, 0, stream>>>(wo, WOb, (int)(WSQ / 4));
  cvt_f32_bf16<<<dim3((int)(WFF / 4 / 256)), blk, 0, stream>>>(w1, W1b, (int)(WFF / 4));
  cvt_f32_bf16<<<dim3((int)(WFF / 4 / 256)), blk, 0, stream>>>(w2, W2b, (int)(WFF / 4));

  // ---- Q/K/V projections (bf16 outputs for attention)
  gemm_bf16<false, bf16_t><<<dim3(D_MODEL / 128, ROWS / 64), blk, 0, stream>>>(
      x, WQb, bq, Qbf, ROWS, D_MODEL, D_MODEL);
  gemm_bf16<false, bf16_t><<<dim3(D_MODEL / 128, ROWS / 64), blk, 0, stream>>>(
      x, WKb, bk, Kbf, ROWS, D_MODEL, D_MODEL);
  gemm_bf16<false, bf16_t><<<dim3(D_MODEL / 128, ROWS / 64), blk, 0, stream>>>(
      x, WVb, bv, Vbf, ROWS, D_MODEL, D_MODEL);

  // ---- causal attention: 512 blocks, each = 8 q-tiles of one (b,h)
  attention_causal<<<dim3(BATCH * N_HEADS * (SEQ / 16) / 8), blk, 0, stream>>>(
      Qbf, Kbf, Vbf, CTX);

  // ---- output projection (fp32 out for residual)
  gemm_bf16<false, float><<<dim3(D_MODEL / 128, ROWS / 64), blk, 0, stream>>>(
      CTX, WOb, bo, ATT, ROWS, D_MODEL, D_MODEL);

  // ---- out1 = LN(x + attn_out)
  add_layernorm<<<dim3(ROWS), blk, 0, stream>>>(x, ATT, g1, be1, O1);

  // ---- h1 = relu(out1 @ w1 + b1)
  gemm_bf16<true, float><<<dim3(DFF / 128, ROWS / 64), blk, 0, stream>>>(
      O1, W1b, b1, H1, ROWS, DFF, D_MODEL);

  // ---- ffn = h1 @ w2 + b2
  gemm_bf16<false, float><<<dim3(D_MODEL / 128, ROWS / 64), blk, 0, stream>>>(
      H1, W2b, b2, FFN, ROWS, D_MODEL, DFF);

  // ---- out2 = LN(out1 + ffn)
  add_layernorm<<<dim3(ROWS), blk, 0, stream>>>(O1, FFN, g2, be2, out);

  (void)in_sizes; (void)n_in; (void)out_size; (void)ws_size;
}
